// MambaBlock_8847632629836
// MI455X (gfx1250) — compile-verified, matrix operand order checked by reading, not run on hardware
//
#include <hip/hip_runtime.h>
#include <hip/hip_bf16.h>

// ---------------------------------------------------------------------------
// Mamba block forward for MI455X (gfx1250), wave32 + WMMA bf16 + TDM.
//   BS=1, L=2048, D_MODEL=512, D_INNER=1024, D_STATE=32, D_CONV=4
// Pipeline: GEMM(WMMA, TDM-staged A, K-unroll 2) -> conv+SiLU -> x_dbc ->
//           selective scan -> GEMM(WMMA)+residual -> LayerNorm
// ---------------------------------------------------------------------------

typedef __bf16        v16bf __attribute__((ext_vector_type(16)));
typedef float         v8f   __attribute__((ext_vector_type(8)));
typedef unsigned int  v4u   __attribute__((ext_vector_type(4)));
typedef int           v4i   __attribute__((ext_vector_type(4)));
typedef int           v8i   __attribute__((ext_vector_type(8)));

#define D_MODEL 512
#define D_STATE 32
#define D_CONV  4
#define D_INNER 1024
#define SEQ_L   2048
#define LN_EPS  1e-5f

#if defined(__gfx1250__) && __has_builtin(__builtin_amdgcn_tensor_load_to_lds)
#define USE_TDM 1
#endif

__device__ __forceinline__ float silu_f(float v) { return v / (1.0f + __expf(-v)); }

// ---------------------------------------------------------------------------
// Tiled bf16 WMMA GEMM:  C[M,N] = A[M,K] @ B[K,N]  (+ optional residual)
// Block = 128 threads = 4 waves; block tile 16(M) x 64(N), one 16x16 WMMA
// tile per wave, K-chunk = 64 (two 16x16x32 WMMAs back-to-back per barrier).
// A tile (16x64 fp32) lands in LDS via one Tensor Data Mover DMA
// (tensor_load_to_lds + s_wait_tensorcnt), then both A and B are converted
// to bf16 in *fragment order* so each lane's v16bf operand is one contiguous
// 32-byte run (2x ds_load_b128).
//
// Fragment K-map (ISA 7.12.2, 16-bit A 16x32 / symmetric B):
//   lane<16:  e=0..7 -> K=0..7,   e=8..15 -> K=16..23
//   lane>=16: e=0..7 -> K=8..15,  e=8..15 -> K=24..31
// ---------------------------------------------------------------------------
template <bool HAS_RES>
__global__ __launch_bounds__(128) void gemm_bf16_wmma(
    const float* __restrict__ A, const float* __restrict__ B,
    const float* __restrict__ resid, float* __restrict__ C,
    int M, int N, int K, int lda, int ldb, int ldc)
{
  __shared__ float Af32[16 * 64];                     // raw A tile (TDM dest)
  __shared__ alignas(32) __bf16 Asf[2][32][16];       // fragment-ordered A
  __shared__ alignas(32) __bf16 Bsf[2][4][32][16];    // fragment-ordered B

  const int tid  = threadIdx.x;
  const int wave = tid >> 5;
  const int lane = tid & 31;
  const int hi   = (lane >= 16) ? 8 : 0;
  const int mrow = lane & 15;
  const int m0   = blockIdx.y * 16;
  const int nb   = blockIdx.x * 64;
  const int n0   = nb + wave * 16;

  v8f acc = {};

  for (int kk = 0; kk < K; kk += 64) {
    // ---- stage A tile (16x64 fp32) into LDS via TDM ------------------
#ifdef USE_TDM
    if (wave == 0) {
      // Tensor DMA Descriptor (ISA ch.8): 2-D tile 64(K) x 16(M), 4B elems.
      unsigned ldsoff = (unsigned)(unsigned long long)(uintptr_t)(&Af32[0]);
      unsigned long long ga =
          (unsigned long long)(uintptr_t)(A + (size_t)m0 * lda + kk);
      v4u g0;
      g0[0] = 1u;                                   // count=1, user D#
      g0[1] = ldsoff;                               // lds_addr (bytes)
      g0[2] = (unsigned)ga;                         // global_addr[31:0]
      g0[3] = (unsigned)((ga >> 32) & 0x01FFFFFFu)  // global_addr[56:32]
              | (2u << 30);                         // type=2 ("image")
      v8i g1;
      g1[0] = (int)(2u << 16);                      // wg_mask=0, data_size=4B
      g1[1] = (int)(((unsigned)lda & 0xFFFFu) << 16);          // dim0[15:0]
      g1[2] = (int)(((unsigned)lda >> 16) |
                    (((unsigned)M & 0xFFFFu) << 16));          // dim0 hi|dim1 lo
      g1[3] = (int)(((unsigned)M >> 16) | (64u << 16));        // dim1 hi|tile0=64
      g1[4] = 16;                                   // tile_dim1=16, tile_dim2=0
      g1[5] = (int)lda;                             // dim0_stride[31:0]
      g1[6] = 0;
      g1[7] = 0;
      v4i gz = {0, 0, 0, 0};
#if __clang_major__ >= 23
      v8i gz8 = {0, 0, 0, 0, 0, 0, 0, 0};
      __builtin_amdgcn_tensor_load_to_lds(g0, g1, gz, gz, gz8, 0);
#else
      __builtin_amdgcn_tensor_load_to_lds(g0, g1, gz, gz, 0);
#endif
      __builtin_amdgcn_s_wait_tensorcnt(0);         // TENSORcnt is per-wave
    }
#else
#pragma unroll
    for (int j = 0; j < (16 * 64) / 128; ++j) {
      int i = tid + j * 128;
      int r = i >> 6, c = i & 63;
      Af32[r * 64 + c] = A[(m0 + r) * lda + kk + c];
    }
#endif

    // ---- stage B tile (64x64) directly into fragment order -----------
    // Unrolled so all 32 global_load_b32 issue before any wait.
#pragma unroll
    for (int j = 0; j < (64 * 64) / 128; ++j) {
      int i  = tid + j * 128;
      int r  = i >> 6, c = i & 63;                  // r=k in chunk pair, c=n
      int kc = r >> 5, rr = r & 31;
      int w  = c >> 4;
      int ln = (c & 15) + ((rr & 8) ? 16 : 0);
      int e  = (rr & 7) + ((rr & 16) ? 8 : 0);
      Bsf[kc][w][ln][e] = (__bf16)B[(kk + r) * ldb + nb + c];
    }
    __builtin_prefetch(&B[(kk + 64) * ldb + nb], 0, 0);  // global_prefetch_b8
    __syncthreads();                                // Af32 + Bsf published

    // ---- convert A tile into fragment order --------------------------
#pragma unroll
    for (int j = 0; j < (16 * 64) / 128; ++j) {
      int i  = tid + j * 128;
      int r  = i >> 6, c = i & 63;                  // r=m, c=k in chunk pair
      int kc = c >> 5, cc = c & 31;
      int ln = r + ((cc & 8) ? 16 : 0);
      int e  = (cc & 7) + ((cc & 16) ? 8 : 0);
      Asf[kc][ln][e] = (__bf16)Af32[r * 64 + c];
    }
    __syncthreads();                                // Asf published

    // ---- two back-to-back WMMAs (accumulator chaining, D->C) ---------
    v16bf a0 = *(const v16bf*)(&Asf[0][lane][0]);
    v16bf b0 = *(const v16bf*)(&Bsf[0][wave][lane][0]);
    v16bf a1 = *(const v16bf*)(&Asf[1][lane][0]);
    v16bf b1 = *(const v16bf*)(&Bsf[1][wave][lane][0]);
    acc = __builtin_amdgcn_wmma_f32_16x16x32_bf16(false, a0, false, b0,
                                                  (short)0, acc, false, false);
    acc = __builtin_amdgcn_wmma_f32_16x16x32_bf16(false, a1, false, b1,
                                                  (short)0, acc, false, false);
    __syncthreads();                                // protect LDS reuse
  }

  // C/D layout: VGPR r -> M = r + 8*(lane>=16), N = lane&15
#pragma unroll
  for (int r = 0; r < 8; ++r) {
    int m = m0 + r + hi;
    int n = n0 + mrow;
    float v = acc[r];
    if (HAS_RES) v += resid[m * ldc + n];
    C[m * ldc + n] = v;
  }
}

// ---------------------------------------------------------------------------
// Depthwise causal conv (width 4) + bias + SiLU over u = xz[:, :D_INNER]
// ---------------------------------------------------------------------------
__global__ void conv_silu_kernel(const float* __restrict__ xz,
                                 const float* __restrict__ cw,
                                 const float* __restrict__ cb,
                                 float* __restrict__ uc)
{
  int idx = blockIdx.x * blockDim.x + threadIdx.x;
  if (idx >= SEQ_L * D_INNER) return;
  int l = idx >> 10;           // / D_INNER
  int d = idx & (D_INNER - 1);
  float s = cb[d];
#pragma unroll
  for (int j = 0; j < D_CONV; ++j) {
    int t = l + j - (D_CONV - 1);
    float uv = (t >= 0) ? xz[t * (2 * D_INNER) + d] : 0.0f;
    s += uv * cw[d * D_CONV + j];
  }
  uc[idx] = silu_f(s);
}

// ---------------------------------------------------------------------------
// x_dbc = u_conv @ W_x  (N=65: too skinny for WMMA tiles; LDS row-dot)
// ---------------------------------------------------------------------------
__global__ __launch_bounds__(128) void xdbc_kernel(const float* __restrict__ uc,
                                                   const float* __restrict__ Wx,
                                                   float* __restrict__ xdbc)
{
  __shared__ float row[D_INNER];
  int l = blockIdx.x;
  for (int i = threadIdx.x; i < D_INNER; i += 128)
    row[i] = uc[l * D_INNER + i];
  __syncthreads();
  for (int c = threadIdx.x; c < (2 * D_STATE + 1); c += 128) {
    float acc = 0.f;
#pragma unroll 4
    for (int k = 0; k < D_INNER; ++k)
      acc += row[k] * Wx[k * (2 * D_STATE + 1) + c];
    xdbc[l * (2 * D_STATE + 1) + c] = acc;
  }
}

// ---------------------------------------------------------------------------
// Selective scan: one wave per channel d, one lane per state s.
// h[d,s] register-resident across L; y contraction = wave32 xor-reduce.
// ---------------------------------------------------------------------------
__global__ __launch_bounds__(256) void scan_kernel(
    const float* __restrict__ uc,   const float* __restrict__ xz,
    const float* __restrict__ xdbc, const float* __restrict__ Wdt,
    const float* __restrict__ bdt,  const float* __restrict__ Alog,
    const float* __restrict__ Dp,   float* __restrict__ y)
{
  const int d = blockIdx.x * 8 + (threadIdx.x >> 5);
  const int s = threadIdx.x & 31;

  const float A_s  = -__expf(Alog[d * D_STATE + s]);
  const float wdt  = Wdt[d];
  const float bdtv = bdt[d];
  const float Dd   = Dp[d];

  float h = 0.f;
  for (int t = 0; t < SEQ_L; ++t) {
    const float u   = uc[t * D_INNER + d];
    const float z   = xz[t * (2 * D_INNER) + D_INNER + d];
    const float dtr = xdbc[t * (2 * D_STATE + 1) + 2 * D_STATE];
    float dt = dtr * wdt + bdtv;
    dt = (dt > 20.f) ? dt : __logf(1.f + __expf(dt));     // softplus
    const float Bv = xdbc[t * (2 * D_STATE + 1) + s];
    const float Cv = xdbc[t * (2 * D_STATE + 1) + D_STATE + s];
    const float dA = __expf(dt * A_s);
    h = dA * h + dt * Bv * u;

    float p = h * Cv;
#pragma unroll
    for (int off = 16; off >= 1; off >>= 1)
      p += __shfl_xor(p, off, 32);                        // wave32 reduce

    if (s == 0) {
      float yv = p + u * Dd;
      y[t * D_INNER + d] = yv * silu_f(z);
    }
  }
}

// ---------------------------------------------------------------------------
// LayerNorm over D_MODEL=512, one block (256 thr) per row.
// ---------------------------------------------------------------------------
__global__ __launch_bounds__(256) void ln_kernel(const float* __restrict__ o,
                                                 const float* __restrict__ g,
                                                 const float* __restrict__ b,
                                                 float* __restrict__ out)
{
  __shared__ float red[256];
  const int l = blockIdx.x, tid = threadIdx.x;
  const float v0 = o[l * D_MODEL + tid];
  const float v1 = o[l * D_MODEL + 256 + tid];

  red[tid] = v0 + v1;
  __syncthreads();
  for (int st = 128; st > 0; st >>= 1) {
    if (tid < st) red[tid] += red[tid + st];
    __syncthreads();
  }
  const float mu = red[0] * (1.0f / D_MODEL);
  __syncthreads();

  const float d0 = v0 - mu, d1 = v1 - mu;
  red[tid] = d0 * d0 + d1 * d1;
  __syncthreads();
  for (int st = 128; st > 0; st >>= 1) {
    if (tid < st) red[tid] += red[tid + st];
    __syncthreads();
  }
  const float rs = rsqrtf(red[0] * (1.0f / D_MODEL) + LN_EPS);

  out[l * D_MODEL + tid]       = d0 * rs * g[tid]       + b[tid];
  out[l * D_MODEL + 256 + tid] = d1 * rs * g[256 + tid] + b[256 + tid];
}

// ---------------------------------------------------------------------------
// launch
// ---------------------------------------------------------------------------
extern "C" void kernel_launch(void* const* d_in, const int* in_sizes, int n_in,
                              void* d_out, int out_size, void* d_ws, size_t ws_size,
                              hipStream_t stream)
{
  const float* x      = (const float*)d_in[0];   // (1,2048,512)
  const float* W_in   = (const float*)d_in[1];   // (512,2048)
  const float* conv_w = (const float*)d_in[2];   // (1024,1,4)
  const float* conv_b = (const float*)d_in[3];   // (1024,)
  const float* W_x    = (const float*)d_in[4];   // (1024,65)
  const float* W_dt   = (const float*)d_in[5];   // (1,1024)
  const float* b_dt   = (const float*)d_in[6];   // (1024,)
  const float* A_log  = (const float*)d_in[7];   // (1024,32)
  const float* Dp     = (const float*)d_in[8];   // (1024,)
  const float* W_out  = (const float*)d_in[9];   // (1024,512)
  const float* ln_g   = (const float*)d_in[10];  // (512,)
  const float* ln_b   = (const float*)d_in[11];  // (512,)
  float* out = (float*)d_out;                    // (1,2048,512)

  // workspace layout (fp32)
  char* ws = (char*)d_ws;
  float* xz    = (float*)(ws);                                   // 2048x2048 : 16 MB
  float* ucnv  = (float*)(ws + (size_t)16 * 1024 * 1024);        // 2048x1024 :  8 MB
  float* xdbc  = (float*)(ws + (size_t)24 * 1024 * 1024);        // 2048x65   : .5 MB
  float* yv    = (float*)(ws + (size_t)25 * 1024 * 1024);        // 2048x1024 :  8 MB
  float* ov    = (float*)(ws + (size_t)33 * 1024 * 1024);        // 2048x512  :  4 MB

  // 1) xz = x @ W_in   (M=2048, K=512, N=2048)
  {
    dim3 grid(2 * D_INNER / 64, SEQ_L / 16);
    gemm_bf16_wmma<false><<<grid, 128, 0, stream>>>(x, W_in, nullptr, xz,
                                                    SEQ_L, 2 * D_INNER, D_MODEL,
                                                    D_MODEL, 2 * D_INNER, 2 * D_INNER);
  }
  // 2) u_conv = SiLU(depthwise_conv(u) + conv_b)
  conv_silu_kernel<<<(SEQ_L * D_INNER) / 256, 256, 0, stream>>>(xz, conv_w, conv_b, ucnv);

  // 3) x_dbc = u_conv @ W_x
  xdbc_kernel<<<SEQ_L, 128, 0, stream>>>(ucnv, W_x, xdbc);

  // 4) selective scan -> y (fused dt/softplus/dA/recurrence/gate)
  scan_kernel<<<D_INNER / 8, 256, 0, stream>>>(ucnv, xz, xdbc, W_dt, b_dt,
                                               A_log, Dp, yv);

  // 5) o = y @ W_out + residual   (M=2048, K=1024, N=512)
  {
    dim3 grid(D_MODEL / 64, SEQ_L / 16);
    gemm_bf16_wmma<true><<<grid, 128, 0, stream>>>(yv, W_out, x, ov,
                                                   SEQ_L, D_MODEL, D_INNER,
                                                   D_INNER, D_MODEL, D_MODEL);
  }
  // 6) LayerNorm -> d_out
  ln_kernel<<<SEQ_L, 256, 0, stream>>>(ov, ln_g, ln_b, out);
}
